// WavRNN_75943611728555
// MI455X (gfx1250) — compile-verified, matrix operand order checked by reading, not run on hardware
//
#include <hip/hip_runtime.h>
#include <hip/hip_bf16.h>

typedef __attribute__((ext_vector_type(16))) __bf16 v16bf;
typedef __attribute__((ext_vector_type(8)))  float  v8f;

#define B_    32
#define T_    2000
#define H_    896
#define Q_    256
#define SUB_  448
#define KSTEP 32
#define NKRNN (H_ / KSTEP)   // 28
#define NKSUB (SUB_ / KSTEP) // 14
#define RNN_BLOCKS 56        // 56 * 16 = 896 output columns per gate

__device__ __forceinline__ unsigned short f2bf(float f) {
    unsigned u = __float_as_uint(f);
    unsigned r = 0x7FFFu + ((u >> 16) & 1u);
    return (unsigned short)((u + r) >> 16);
}

// 16x16x32 bf16 fragment: VGPRs 0-3 = K[half*8 .. +7], VGPRs 4-7 = K[16+half*8 .. +7]
// -> two contiguous 16-byte loads per lane from a row-major [row][k] tile.
union Frag { v16bf v; uint4 q[2]; };

__device__ __forceinline__ v16bf load_frag_lds(const unsigned short* base,
                                               int row, int rowstride,
                                               int k0, int half) {
    Frag f;
    const unsigned short* p = base + row * rowstride + k0 + half * 8;
    f.q[0] = *(const uint4*)(p);
    f.q[1] = *(const uint4*)(p + 16);
    return f.v;
}

__device__ __forceinline__ v16bf load_frag_glb(const unsigned short* __restrict__ base,
                                               int row, int rowstride,
                                               int k0, int half) {
    Frag f;
    const unsigned short* p = base + row * rowstride + k0 + half * 8;
    f.q[0] = *(const uint4*)(p);
    f.q[1] = *(const uint4*)(p + 16);
    return f.v;
}

// Async 16B copy global -> LDS (CDNA5 GLOBAL_LOAD_ASYNC_TO_LDS_B128, ASYNCcnt).
__device__ __forceinline__ void async_copy_b128(unsigned lds_addr, const void* gaddr) {
    asm volatile("global_load_async_to_lds_b128 %0, %1, off"
                 :: "v"(lds_addr), "v"(gaddr) : "memory");
}
__device__ __forceinline__ void async_wait0() {
    asm volatile("s_wait_asynccnt 0x0" ::: "memory");
}

// ---------------------------------------------------------------------------
// Kernel 1: convert f32 -> bf16 for initial state and head weights
// ---------------------------------------------------------------------------
__global__ void wavrnn_convert_kernel(
    const float* __restrict__ state,
    const float* __restrict__ Wc1, const float* __restrict__ Wc2,
    const float* __restrict__ Wf1, const float* __restrict__ Wf2,
    unsigned short* __restrict__ h0, float* __restrict__ hbuf0,
    unsigned short* __restrict__ wc1b, unsigned short* __restrict__ wc2b,
    unsigned short* __restrict__ wf1b, unsigned short* __restrict__ wf2b)
{
    int gid = blockIdx.x * blockDim.x + threadIdx.x;
    int stride = gridDim.x * blockDim.x;
    for (int i = gid; i < B_ * H_; i += stride) {
        h0[i] = f2bf(state[i]);
        hbuf0[i] = state[i];
    }
    for (int i = gid; i < SUB_ * SUB_; i += stride) {
        wc1b[i] = f2bf(Wc1[i]);
        wf1b[i] = f2bf(Wf1[i]);
    }
    for (int i = gid; i < Q_ * SUB_; i += stride) {
        wc2b[i] = f2bf(Wc2[i]);
        wf2b[i] = f2bf(Wf2[i]);
    }
}

// ---------------------------------------------------------------------------
// Kernel 2: persistent recurrent scan. 56 workgroups, each owns 16 columns of
// each of the 3 gates; R slice (bf16) pinned in LDS for all 2000 steps.
// ---------------------------------------------------------------------------
__global__ __launch_bounds__(256, 1)
void wavrnn_scan_kernel(
    const int*   __restrict__ sample, const int* __restrict__ cts,
    const float* __restrict__ R,
    const float* __restrict__ Wc, const float* __restrict__ Wf,
    const float* __restrict__ bu, const float* __restrict__ br,
    const float* __restrict__ be,
    float* __restrict__ hbuf,                    // [2][B_*H_] f32 ping-pong
    const unsigned short* __restrict__ h0,       // [B_*H_] bf16 initial state
    unsigned short* __restrict__ states,         // [T_][B_][H_] bf16
    float* __restrict__ out_last,                // [B_*H_] f32
    unsigned* __restrict__ counter)
{
    extern __shared__ char smem[];
    unsigned short* lR   = (unsigned short*)smem;                  // 48*896*2 = 86016 B
    unsigned short* lH   = (unsigned short*)(smem + 86016);        // 32*896*2 = 57344 B
    float*          lAcc = (float*)(smem + 86016 + 57344);         // 3*32*16*4 = 6144 B
    float*          lFW  = (float*)(smem + 86016 + 57344 + 6144);  // 3*16*4 f32 = 768 B
    float*          lBia = (float*)(smem + 86016 + 57344 + 6144 + 768); // 48 f32 = 192 B

    const int tid = threadIdx.x;
    const int jc  = blockIdx.x * 16;   // column base of this WG's slice
    const unsigned lH_base = (unsigned)(uintptr_t)lH;  // LDS byte offset (low 32b of flat)

    // Load + convert this WG's R slice (rows jc..jc+15 of each gate) to LDS bf16.
    for (int idx = tid; idx < 48 * H_; idx += 256) {
        int row = idx / H_, k = idx - row * H_;
        int g = row >> 4, n = row & 15;
        lR[idx] = f2bf(R[(size_t)(g * H_ + jc + n) * H_ + k]);
    }
    // Cache this WG's input-projection weights + biases (used every timestep).
    if (tid < 48) {
        const int g = tid >> 4, n = tid & 15;
        const int j = jc + n;
        float w0, w1, w2;
        if (j < SUB_) {
            w0 = Wc[(g * SUB_ + j) * 2 + 0];
            w1 = Wc[(g * SUB_ + j) * 2 + 1];
            w2 = 0.f;
        } else {
            const int jj = j - SUB_;
            w0 = Wf[(g * SUB_ + jj) * 3 + 0];
            w1 = Wf[(g * SUB_ + jj) * 3 + 1];
            w2 = Wf[(g * SUB_ + jj) * 3 + 2];
        }
        lFW[tid * 4 + 0] = w0;
        lFW[tid * 4 + 1] = w1;
        lFW[tid * 4 + 2] = w2;
        lBia[tid] = (g == 0 ? bu : (g == 1 ? br : be))[j];
    }

    const int lane = tid & 31;
    const int wave = tid >> 5;
    const int half = lane >> 4;
    const int mrow = lane & 15;

    for (int t = 0; t < T_; ++t) {
        // ---- stage h (bf16) into LDS via async direct-to-LDS copies ----
        const unsigned short* hsrc = (t == 0) ? h0 : (states + (size_t)(t - 1) * B_ * H_);
        for (int i = tid; i < (B_ * H_ * 2) / 16; i += 256)
            async_copy_b128(lH_base + (unsigned)i * 16u, (const char*)hsrc + (size_t)i * 16);
        async_wait0();
        __syncthreads();

        // ---- WMMA: waves 0..5 each own (gate, mtile) ----
        if (wave < 6) {
            const int g  = wave >> 1;
            const int mt = wave & 1;
            const int m  = mt * 16 + mrow;
            const int nrow = g * 16 + (lane & 15);
            v8f acc = {};
#pragma unroll 4
            for (int ks = 0; ks < NKRNN; ++ks) {
                const int k0 = ks * KSTEP;
                v16bf a = load_frag_lds(lH, m, H_, k0, half);
                v16bf b = load_frag_lds(lR, nrow, H_, k0, half);
                acc = __builtin_amdgcn_wmma_f32_16x16x32_bf16(
                    false, a, false, b, (short)0, acc, false, false);
            }
#pragma unroll
            for (int v = 0; v < 8; ++v) {
                int mm = mt * 16 + half * 8 + v;
                lAcc[(g * B_ + mm) * 16 + (lane & 15)] = acc[v];
            }
        }
        __syncthreads();

        // ---- fused gating over the 32x16 slice ----
        const int cur = t & 1;
        const float* hc = hbuf + cur * (B_ * H_);
        float*       hn = hbuf + (cur ^ 1) * (B_ * H_);
        for (int e = tid; e < B_ * 16; e += 256) {
            const int b = e >> 4, n = e & 15;
            const int j = jc + n;
            const float s0 = 2.f * (float)sample[(b * T_ + t) * 2 + 0] * (1.f / 255.f) - 1.f;
            const float s1 = 2.f * (float)sample[(b * T_ + t) * 2 + 1] * (1.f / 255.f) - 1.f;
            const float c0 = 2.f * (float)cts[b * T_ + t] * (1.f / 255.f) - 1.f;
            const float fu = lFW[(0 * 16 + n) * 4 + 0] * s0 + lFW[(0 * 16 + n) * 4 + 1] * s1
                           + lFW[(0 * 16 + n) * 4 + 2] * c0;
            const float fr = lFW[(1 * 16 + n) * 4 + 0] * s0 + lFW[(1 * 16 + n) * 4 + 1] * s1
                           + lFW[(1 * 16 + n) * 4 + 2] * c0;
            const float fe = lFW[(2 * 16 + n) * 4 + 0] * s0 + lFW[(2 * 16 + n) * 4 + 1] * s1
                           + lFW[(2 * 16 + n) * 4 + 2] * c0;
            const float ru = lAcc[(0 * B_ + b) * 16 + n];
            const float rr = lAcc[(1 * B_ + b) * 16 + n];
            const float re = lAcc[(2 * B_ + b) * 16 + n];
            const float ut = 1.f / (1.f + __expf(-(ru + fu + lBia[0 * 16 + n])));
            const float rt = 1.f / (1.f + __expf(-(rr + fr + lBia[1 * 16 + n])));
            const float et = tanhf(rt * re + fe + lBia[2 * 16 + n]);
            const float h_old = hc[b * H_ + j];
            const float h_new = ut * h_old + (1.f - ut) * et;
            hn[b * H_ + j] = h_new;
            states[((size_t)t * B_ + b) * H_ + j] = f2bf(h_new);
            if (t == T_ - 1) out_last[b * H_ + j] = h_new;
        }

        // ---- device-scope grid barrier (monotone counter, graph-safe) ----
        __threadfence();
        __syncthreads();
        if (tid == 0) {
            __hip_atomic_fetch_add(counter, 1u, __ATOMIC_ACQ_REL, __HIP_MEMORY_SCOPE_AGENT);
            const unsigned target = (unsigned)gridDim.x * (unsigned)(t + 1);
            while (__hip_atomic_load(counter, __ATOMIC_ACQUIRE, __HIP_MEMORY_SCOPE_AGENT) < target)
                __builtin_amdgcn_s_sleep(1);
        }
        __syncthreads();
    }
}

// ---------------------------------------------------------------------------
// Kernel 3: output heads. One workgroup per (head, timestep): 32 rows.
// X[32x448] -> relu(X@W1^T + b1) -> @W2^T + b2 -> log_softmax over 256.
// ---------------------------------------------------------------------------
__global__ __launch_bounds__(256, 1)
void wavrnn_heads_kernel(
    const unsigned short* __restrict__ states,
    const unsigned short* __restrict__ wc1b, const unsigned short* __restrict__ wc2b,
    const unsigned short* __restrict__ wf1b, const unsigned short* __restrict__ wf2b,
    const float* __restrict__ bc1, const float* __restrict__ bc2,
    const float* __restrict__ bf1, const float* __restrict__ bf2,
    float* __restrict__ out)
{
    extern __shared__ char smem[];
    unsigned short* lX = (unsigned short*)smem;               // 32*448*2 = 28672 B
    unsigned short* lY = (unsigned short*)(smem + 28672);     // 28672 B
    float*          lZ = (float*)(smem + 57344);              // 32*256*4 = 32768 B

    const int tid  = threadIdx.x;
    const int head = blockIdx.x / T_;
    const int t    = blockIdx.x % T_;
    const unsigned short* W1 = head ? wf1b : wc1b;
    const unsigned short* W2 = head ? wf2b : wc2b;
    const float* b1 = head ? bf1 : bc1;
    const float* b2 = head ? bf2 : bc2;
    float* outh = out + (size_t)head * B_ * T_ * Q_;
    const int hoff = head ? SUB_ : 0;
    const unsigned lX_base = (unsigned)(uintptr_t)lX;

    // stage X (32 rows x 448 bf16) into LDS via async direct-to-LDS copies
    {
        const int rowv = (SUB_ * 2) / 16;   // 56 x uint4 per row
        for (int i = tid; i < B_ * rowv; i += 256) {
            int r = i / rowv, c = i - r * rowv;
            const unsigned short* src = states + ((size_t)t * B_ + r) * H_ + hoff + c * 8;
            async_copy_b128(lX_base + (unsigned)(r * rowv + c) * 16u, src);
        }
        async_wait0();
    }
    __syncthreads();

    const int lane = tid & 31, wave = tid >> 5;
    const int half = lane >> 4, mrow = lane & 15, nlane = lane & 15;

    // GEMM1: Y = relu(X @ W1^T + b1), 56 tiles over 8 waves
    for (int tile = wave; tile < 56; tile += 8) {
        const int mt = tile & 1, nt = tile >> 1;
        const int m    = mt * 16 + mrow;
        const int orow = nt * 16 + nlane;
        v8f acc = {};
#pragma unroll 2
        for (int ks = 0; ks < NKSUB; ++ks) {
            const int k0 = ks * KSTEP;
            __builtin_prefetch(W1 + orow * SUB_ + k0 + 64, 0, 1);
            v16bf a = load_frag_lds(lX, m, SUB_, k0, half);
            v16bf b = load_frag_glb(W1, orow, SUB_, k0, half);
            acc = __builtin_amdgcn_wmma_f32_16x16x32_bf16(
                false, a, false, b, (short)0, acc, false, false);
        }
        const int col = nt * 16 + nlane;
#pragma unroll
        for (int v = 0; v < 8; ++v) {
            const int mm = mt * 16 + half * 8 + v;
            float y = acc[v] + b1[col];
            lY[mm * SUB_ + col] = f2bf(fmaxf(y, 0.f));
        }
    }
    __syncthreads();

    // GEMM2: Z = Y @ W2^T + b2, 32 tiles over 8 waves
    for (int tile = wave; tile < 32; tile += 8) {
        const int mt = tile & 1, nt = tile >> 1;
        const int m    = mt * 16 + mrow;
        const int orow = nt * 16 + nlane;
        v8f acc = {};
#pragma unroll 2
        for (int ks = 0; ks < NKSUB; ++ks) {
            const int k0 = ks * KSTEP;
            __builtin_prefetch(W2 + orow * SUB_ + k0 + 64, 0, 1);
            v16bf a = load_frag_lds(lY, m, SUB_, k0, half);
            v16bf b = load_frag_glb(W2, orow, SUB_, k0, half);
            acc = __builtin_amdgcn_wmma_f32_16x16x32_bf16(
                false, a, false, b, (short)0, acc, false, false);
        }
        const int col = nt * 16 + nlane;
#pragma unroll
        for (int v = 0; v < 8; ++v) {
            const int mm = mt * 16 + half * 8 + v;
            lZ[mm * Q_ + col] = acc[v] + b2[col];
        }
    }
    __syncthreads();

    // log_softmax over Q=256 per row; 8 waves x 4 rows, 8 cols per lane
    for (int q = 0; q < 4; ++q) {
        const int m = wave * 4 + q;   // row == batch index b
        float zv[8];
        float mx = -1e30f;
#pragma unroll
        for (int i = 0; i < 8; ++i) {
            zv[i] = lZ[m * Q_ + lane + 32 * i];
            mx = fmaxf(mx, zv[i]);
        }
#pragma unroll
        for (int off = 16; off > 0; off >>= 1)
            mx = fmaxf(mx, __shfl_xor(mx, off, 32));
        float s = 0.f;
#pragma unroll
        for (int i = 0; i < 8; ++i) s += __expf(zv[i] - mx);
#pragma unroll
        for (int off = 16; off > 0; off >>= 1)
            s += __shfl_xor(s, off, 32);
        const float lse = mx + __logf(s);
        const size_t base = ((size_t)m * T_ + t) * Q_;
#pragma unroll
        for (int i = 0; i < 8; ++i)
            outh[base + lane + 32 * i] = zv[i] - lse;
    }
}

// ---------------------------------------------------------------------------
extern "C" void kernel_launch(void* const* d_in, const int* in_sizes, int n_in,
                              void* d_out, int out_size, void* d_ws, size_t ws_size,
                              hipStream_t stream)
{
    const int*   sample = (const int*)d_in[0];
    const int*   cts    = (const int*)d_in[1];
    const float* state  = (const float*)d_in[2];
    const float* R      = (const float*)d_in[3];
    const float* Wc     = (const float*)d_in[4];
    const float* Wf     = (const float*)d_in[5];
    const float* bu     = (const float*)d_in[6];
    const float* br     = (const float*)d_in[7];
    const float* be     = (const float*)d_in[8];
    const float* Wc1    = (const float*)d_in[9];
    const float* bc1    = (const float*)d_in[10];
    const float* Wc2    = (const float*)d_in[11];
    const float* bc2    = (const float*)d_in[12];
    const float* Wf1    = (const float*)d_in[13];
    const float* bf1    = (const float*)d_in[14];
    const float* Wf2    = (const float*)d_in[15];
    const float* bf2    = (const float*)d_in[16];
    float* out = (float*)d_out;

    char* ws = (char*)d_ws;
    // workspace layout (bytes)
    const size_t OFF_CNT    = 0;                                   // 256
    const size_t OFF_HBUF   = 256;                                 // 2*B*H f32 = 229376
    const size_t OFF_H0     = OFF_HBUF + 2ull * B_ * H_ * 4;       // B*H bf16 = 57344
    const size_t OFF_WC1    = OFF_H0 + (size_t)B_ * H_ * 2;        // 448*448*2
    const size_t OFF_WF1    = OFF_WC1 + (size_t)SUB_ * SUB_ * 2;
    const size_t OFF_WC2    = OFF_WF1 + (size_t)SUB_ * SUB_ * 2;   // 256*448*2
    const size_t OFF_WF2    = OFF_WC2 + (size_t)Q_ * SUB_ * 2;
    const size_t OFF_STATES = OFF_WF2 + (size_t)Q_ * SUB_ * 2;     // T*B*H bf16

    unsigned*       counter = (unsigned*)(ws + OFF_CNT);
    float*          hbuf    = (float*)(ws + OFF_HBUF);
    unsigned short* h0      = (unsigned short*)(ws + OFF_H0);
    unsigned short* wc1b    = (unsigned short*)(ws + OFF_WC1);
    unsigned short* wf1b    = (unsigned short*)(ws + OFF_WF1);
    unsigned short* wc2b    = (unsigned short*)(ws + OFF_WC2);
    unsigned short* wf2b    = (unsigned short*)(ws + OFF_WF2);
    unsigned short* states  = (unsigned short*)(ws + OFF_STATES);

    float* out_last = out + 2ull * B_ * T_ * Q_;

    hipMemsetAsync(counter, 0, 256, stream);

    wavrnn_convert_kernel<<<512, 256, 0, stream>>>(
        state, Wc1, Wc2, Wf1, Wf2, h0, hbuf, wc1b, wc2b, wf1b, wf2b);

    const size_t scanLds  = 86016 + 57344 + 6144 + 768 + 192;  // R + h + acc + fw + bias
    wavrnn_scan_kernel<<<RNN_BLOCKS, 256, scanLds, stream>>>(
        sample, cts, R, Wc, Wf, bu, br, be,
        hbuf, h0, states, out_last, counter);

    const size_t headsLds = 28672 + 28672 + 32768;   // X + Y + Z
    wavrnn_heads_kernel<<<2 * T_, 256, headsLds, stream>>>(
        states, wc1b, wc2b, wf1b, wf2b, bc1, bc2, bf1, bf2, out);
}
